// SpanInteractionAttention_63874753626508
// MI455X (gfx1250) — matrix operand order repaired and checked by reading, not compiled, not run on hardware
//
#include <hip/hip_runtime.h>
#include <hip/hip_bf16.h>

typedef __attribute__((ext_vector_type(16))) _Float16 v16h;
typedef __attribute__((ext_vector_type(8)))  _Float16 v8h;
typedef __attribute__((ext_vector_type(8)))  float    v8f;

#define NTOK 2048
#define DMODEL 512
#define DFFN 1024
#define BATCH 8
#define MROWS (BATCH * NTOK)   // 16384

__device__ __forceinline__ v8f wmma_f16(v16h a, v16h b, v8f c) {
    // D = A(16x32 f16) * B(32x16 f16) + C(16x16 f32)
    return __builtin_amdgcn_wmma_f32_16x16x32_f16(
        /*neg_a=*/false, a, /*neg_b=*/false, b,
        /*c_mod=*/(short)0, c, /*reuse_a=*/false, /*reuse_b=*/false);
}

__device__ __forceinline__ v16h combine16(v8h lo, v8h hi) {
    v16h r;
#pragma unroll
    for (int i = 0; i < 8; ++i) { r[i] = lo[i]; r[i + 8] = hi[i]; }
    return r;
}

// ---------------------------------------------------------------------------
// Kernel 1: W[K][N] (f32, row-major) -> Wt[N][K] (f16, row-major)
// ---------------------------------------------------------------------------
__global__ void convert_transpose_kernel(const float* __restrict__ W,
                                         _Float16* __restrict__ Wt,
                                         int K, int N) {
    int idx = blockIdx.x * blockDim.x + threadIdx.x;
    if (idx >= K * N) return;
    int n = idx / K;
    int k = idx - n * K;
    Wt[idx] = (_Float16)W[(size_t)k * N + n];   // coalesced writes
}

// ---------------------------------------------------------------------------
// Kernel 2: tiled WMMA GEMM   C[M][Nc] = act(A[M][K] * Bt[Nc][K]^T + bias)
//   wave -> 32x32 C tile (2x2 register blocking: 4 frag loads -> 4 WMMAs)
//   block = 8 waves (2M x 4N) -> 64(M) x 128(N) block tile
//   A_F32: A is fp32 (converted in-register), else f16
//   SPLIT: route columns to k / q / vT buffers (FFN2), else plain f16 C
// ---------------------------------------------------------------------------
template <bool A_F32, bool RELU, bool SPLIT>
__global__ void __launch_bounds__(256)
gemm_wmma_kernel(const void* __restrict__ Aptr,
                 const _Float16* __restrict__ Bt,
                 const float* __restrict__ bias,
                 _Float16* __restrict__ Cout,
                 _Float16* __restrict__ Kout,
                 _Float16* __restrict__ Qout,
                 _Float16* __restrict__ VTout,
                 int M, int K, int Nc) {
    const int lane = threadIdx.x & 31;
    const int wave = threadIdx.x >> 5;
    const int hl   = lane & 15;      // row (A) / col (B,C) within a 16-tile
    const int g    = lane >> 4;      // lane half-group
    const int wm   = wave >> 2;      // 0..1
    const int wn   = wave & 3;       // 0..3

    const int tileM = blockIdx.y * 64 + wm * 32;
    const int tileN = blockIdx.x * 128 + wn * 32;
    const size_t arow[2] = {(size_t)(tileM + hl), (size_t)(tileM + 16 + hl)};
    const int    col[2]  = {tileN + hl, tileN + 16 + hl};

    v8f acc[2][2] = {{{}, {}}, {{}, {}}};

    for (int k0 = 0; k0 < K; k0 += 32) {
        v16h a[2], bf[2];
#pragma unroll
        for (int mi = 0; mi < 2; ++mi) {
            if (A_F32) {
                const float* Ap = (const float*)Aptr + arow[mi] * K + k0;
#pragma unroll
                for (int i = 0; i < 8; ++i) {
                    a[mi][i]     = (_Float16)Ap[8 * g + i];      // K = 8g..8g+7
                    a[mi][i + 8] = (_Float16)Ap[16 + 8 * g + i]; // K = 16+8g..
                }
            } else {
                const _Float16* Ap = (const _Float16*)Aptr + arow[mi] * K + k0;
                a[mi] = combine16(*(const v8h*)(Ap + 8 * g),
                                  *(const v8h*)(Ap + 16 + 8 * g));
            }
        }
#pragma unroll
        for (int ni = 0; ni < 2; ++ni)
            bf[ni] = combine16(*(const v8h*)(Bt + (size_t)col[ni] * K + k0 + 8 * g),
                               *(const v8h*)(Bt + (size_t)col[ni] * K + k0 + 16 + 8 * g));
#pragma unroll
        for (int mi = 0; mi < 2; ++mi)
#pragma unroll
            for (int ni = 0; ni < 2; ++ni)
                acc[mi][ni] = wmma_f16(a[mi], bf[ni], acc[mi][ni]);
    }

#pragma unroll
    for (int ni = 0; ni < 2; ++ni) {
        const int   c  = col[ni];
        const float bv = bias[c];
#pragma unroll
        for (int mi = 0; mi < 2; ++mi) {
#pragma unroll
            for (int r = 0; r < 8; ++r) {
                float v = acc[mi][ni][r] + bv;
                if (RELU) v = fmaxf(v, 0.0f);
                const _Float16 hv = (_Float16)v;
                const int orow = tileM + mi * 16 + 8 * g + r;  // C-layout row
                if (!SPLIT) {
                    Cout[(size_t)orow * Nc + c] = hv;
                } else {
                    // kqv split: [0,512)=k, [512,1024)=q, [1024,1536)=v^T
                    if (c < DMODEL) {
                        Kout[(size_t)orow * DMODEL + c] = hv;
                    } else if (c < 2 * DMODEL) {
                        Qout[(size_t)orow * DMODEL + (c - DMODEL)] = hv;
                    } else {
                        const int bidx = orow >> 11;          // / NTOK
                        const int tok  = orow & (NTOK - 1);
                        VTout[((size_t)bidx * DMODEL + (c - 2 * DMODEL)) * NTOK
                              + tok] = hv;
                    }
                }
            }
        }
    }
}

// ---------------------------------------------------------------------------
// Kernel 3: fused mask-attention.
//   block: 256 thr (8 waves) owns (b, 16-row q tile), full 512 out cols.
//   q tile staged into LDS with GLOBAL_LOAD_ASYNC_TO_LDS_B128 (ASYNCcnt).
//   per 128-token chunk: each wave computes one 16x16 S tile (QK^T, 16 WMMAs),
//   scale*mask -> f16 -> LDS; then each wave accumulates 4 d-tiles vs vT.
// ---------------------------------------------------------------------------
__global__ void __launch_bounds__(256)
attn_wmma_kernel(const _Float16* __restrict__ Qb,
                 const _Float16* __restrict__ Kb,
                 const _Float16* __restrict__ VT,
                 const float* __restrict__ mask,
                 float* __restrict__ out) {
    __shared__ __align__(16) _Float16 sQ[16 * DMODEL];  // 16 KB
    __shared__ __align__(16) _Float16 sS[16 * 128];     // 4 KB

    const int b   = blockIdx.y;
    const int q0  = blockIdx.x * 16;
    const int tid = threadIdx.x;

    // stage the 16 x 512 q tile once: async global -> LDS, 16B per lane x4
    {
        const _Float16* src = Qb + ((size_t)b * NTOK + q0) * DMODEL;
        const unsigned lds0 = (unsigned)(uintptr_t)(&sQ[0]);
#pragma unroll
        for (int i = 0; i < 4; ++i) {
            unsigned ldsaddr = lds0 + (unsigned)(tid + 256 * i) * 16u;
            unsigned long long gaddr =
                (unsigned long long)(uintptr_t)(src + (tid + 256 * i) * 8);
            asm volatile("global_load_async_to_lds_b128 %0, %1, off"
                         :: "v"(ldsaddr), "v"(gaddr) : "memory");
        }
        asm volatile("s_wait_asynccnt 0x0" ::: "memory");
    }
    __syncthreads();

    const int lane = tid & 31;
    const int wave = tid >> 5;
    const int hl   = lane & 15;
    const int g    = lane >> 4;
    const float scale = 0.04419417382415922f;  // 512^-0.5

    v8f o[4] = {{}, {}, {}, {}};

    for (int kc = 0; kc < NTOK; kc += 128) {
        // ---- phase 1: S[16, kc+16w .. +15] = q . k^T ----
        const int scol = kc + wave * 16 + hl;              // key token index
        const _Float16* Kp = Kb + ((size_t)b * NTOK + scol) * DMODEL;
        v8f s = {};
#pragma unroll 4
        for (int d0 = 0; d0 < DMODEL; d0 += 32) {
            const _Float16* qp = sQ + hl * DMODEL + d0;
            v16h a = combine16(*(const v8h*)(qp + 8 * g),
                               *(const v8h*)(qp + 16 + 8 * g));
            v16h bf = combine16(*(const v8h*)(Kp + d0 + 8 * g),
                                *(const v8h*)(Kp + d0 + 16 + 8 * g));
            s = wmma_f16(a, bf, s);
        }
        const float* mrow = mask + ((size_t)b * NTOK + q0) * NTOK + scol;
#pragma unroll
        for (int r = 0; r < 8; ++r) {
            const int qr = r + 8 * g;
            const float mv = mrow[(size_t)qr * NTOK];
            sS[qr * 128 + wave * 16 + hl] = (_Float16)(s[r] * scale * mv);
        }
        __syncthreads();

        // ---- phase 2: O[16, wave*64 + t*16 + hl] += S . V ----
#pragma unroll
        for (int kk = 0; kk < 128; kk += 32) {
            const _Float16* sp = sS + hl * 128 + kk;
            v16h a = combine16(*(const v8h*)(sp + 8 * g),
                               *(const v8h*)(sp + 16 + 8 * g));
#pragma unroll
            for (int t = 0; t < 4; ++t) {
                const int n = wave * 64 + t * 16 + hl;     // output d column
                const _Float16* vp =
                    VT + ((size_t)b * DMODEL + n) * NTOK + kc + kk;
                v16h bf = combine16(*(const v8h*)(vp + 8 * g),
                                    *(const v8h*)(vp + 16 + 8 * g));
                o[t] = wmma_f16(a, bf, o[t]);
            }
        }
        __syncthreads();
    }

#pragma unroll
    for (int t = 0; t < 4; ++t) {
        const int n = wave * 64 + t * 16 + hl;
#pragma unroll
        for (int r = 0; r < 8; ++r) {
            const int qr = q0 + r + 8 * g;
            out[((size_t)b * NTOK + qr) * DMODEL + n] = o[t][r];
        }
    }
}

// ---------------------------------------------------------------------------
// launch
// ---------------------------------------------------------------------------
extern "C" void kernel_launch(void* const* d_in, const int* in_sizes, int n_in,
                              void* d_out, int out_size, void* d_ws, size_t ws_size,
                              hipStream_t stream) {
    const float* x    = (const float*)d_in[0];
    const float* mask = (const float*)d_in[1];
    const float* W1   = (const float*)d_in[2];
    const float* b1   = (const float*)d_in[3];
    const float* W2   = (const float*)d_in[4];
    const float* b2   = (const float*)d_in[5];
    float* out = (float*)d_out;

    // workspace layout (f16 elements)
    _Float16* ws   = (_Float16*)d_ws;
    _Float16* W1t  = ws;                               // 1024*512      = 0.5M
    _Float16* W2t  = W1t + (size_t)DFFN * DMODEL;      // 1536*1024     = 1.5M
    _Float16* Hbuf = W2t + (size_t)(3 * DMODEL) * DFFN;// 16384*1024    = 16M
    _Float16* Kbuf = Hbuf + (size_t)MROWS * DFFN;      // 16384*512     = 8M
    _Float16* Qbuf = Kbuf + (size_t)MROWS * DMODEL;
    _Float16* VTb  = Qbuf + (size_t)MROWS * DMODEL;    // [B][512][2048]

    // 1) weight convert + transpose to f16
    convert_transpose_kernel<<<(DMODEL * DFFN + 255) / 256, 256, 0, stream>>>(
        W1, W1t, DMODEL, DFFN);
    convert_transpose_kernel<<<(DFFN * 3 * DMODEL + 255) / 256, 256, 0, stream>>>(
        W2, W2t, DFFN, 3 * DMODEL);

    // 2) FFN1: h = relu(x @ W1 + b1)   [16384 x 1024], A fp32
    gemm_wmma_kernel<true, true, false>
        <<<dim3(DFFN / 128, MROWS / 64), 256, 0, stream>>>(
            x, W1t, b1, Hbuf, nullptr, nullptr, nullptr, MROWS, DMODEL, DFFN);

    // 3) FFN2: kqv = h @ W2 + b2, split -> k, q (row-major), v (transposed)
    gemm_wmma_kernel<false, false, true>
        <<<dim3((3 * DMODEL) / 128, MROWS / 64), 256, 0, stream>>>(
            Hbuf, W2t, b2, nullptr, Kbuf, Qbuf, VTb, MROWS, DFFN, 3 * DMODEL);

    // 4) fused (q k^T * scale * mask) @ v
    attn_wmma_kernel<<<dim3(NTOK / 16, BATCH), 256, 0, stream>>>(
        Qbuf, Kbuf, VTb, mask, out);
}
